// RNNDecoder_12034498363583
// MI455X (gfx1250) — matrix operand order repaired and checked by reading, not compiled, not run on hardware
//
#include <hip/hip_runtime.h>
#include <hip/hip_bf16.h>
#include <math.h>
#include <stdint.h>

// ---------------- problem constants ----------------
#define BB   1024          // batch
#define WW   128           // time steps
#define EE   512           // embed size
#define HH   512           // hidden size
#define NN   2048          // 4*H gate width
#define KK   1024          // concat K = E + H
#define PITCH 40           // LDS row pitch in ushorts (80 B) to stagger banks

typedef __attribute__((ext_vector_type(16))) __bf16          v16bf;
typedef __attribute__((ext_vector_type(8)))  float           v8f;
typedef __attribute__((ext_vector_type(4)))  int             v4i;   // native vector, not HIP struct

#ifndef __has_builtin
#define __has_builtin(x) 0
#endif
#if __has_builtin(__builtin_amdgcn_global_load_async_to_lds_b128) && \
    __has_builtin(__builtin_amdgcn_s_wait_asynccnt)
#define USE_ASYNC 1
#else
#define USE_ASYNC 0
#endif

typedef __attribute__((address_space(3))) v4i* as3_v4i;

union V32 { uint4 q[2]; v16bf v; };

// round-to-nearest-even f32 -> bf16 (bit pattern)
static __device__ __forceinline__ unsigned short f2bf(float f) {
    unsigned u = __float_as_uint(f);
    u += 0x7FFFu + ((u >> 16) & 1u);
    return (unsigned short)(u >> 16);
}

static __device__ __forceinline__ float sigm(float x) {
    return 1.0f / (1.0f + __expf(-x));
}

// ---------------- prep: fuse weights to bf16, sum biases ----------------
__global__ void lstm_prep(const float* __restrict__ Wih, const float* __restrict__ Whh,
                          const float* __restrict__ bih, const float* __restrict__ bhh,
                          unsigned short* __restrict__ Wc, float* __restrict__ bias) {
    int idx = blockIdx.x * blockDim.x + threadIdx.x;
    if (idx < NN * KK) {
        int n = idx >> 10;
        int k = idx & (KK - 1);
        float w = (k < EE) ? Wih[n * EE + k] : Whh[n * HH + (k - EE)];
        Wc[idx] = f2bf(w);
    }
    if (idx < NN) bias[idx] = bih[idx] + bhh[idx];
}

// zero cell state and the h-half of the bf16 activation buffer
__global__ void lstm_init(float* __restrict__ C, unsigned short* __restrict__ Act) {
    int idx = blockIdx.x * blockDim.x + threadIdx.x;
    if (idx < BB * HH) {
        C[idx] = 0.0f;
        Act[(size_t)(idx / HH) * KK + EE + (idx % HH)] = 0;  // bf16 +0.0
    }
}

// per-step: convert x_t (f32) into Act[:, 0:512] (bf16); 8 elems / thread
__global__ void lstm_xcast(const float* __restrict__ inp,
                           unsigned short* __restrict__ Act, int t) {
    int idx = blockIdx.x * blockDim.x + threadIdx.x;   // 65536 threads
    int b = idx >> 6;
    int c = (idx & 63) * 8;
    const float* p = inp + (size_t)b * (WW * EE) + (size_t)t * EE + c;
    float4 f0 = ((const float4*)p)[0];
    float4 f1 = ((const float4*)p)[1];
    union { unsigned short s[8]; uint4 q; } u;
    u.s[0] = f2bf(f0.x); u.s[1] = f2bf(f0.y); u.s[2] = f2bf(f0.z); u.s[3] = f2bf(f0.w);
    u.s[4] = f2bf(f1.x); u.s[5] = f2bf(f1.y); u.s[6] = f2bf(f1.z); u.s[7] = f2bf(f1.w);
    *(uint4*)&Act[(size_t)b * KK + c] = u.q;
}

// cooperative fill of one K-slab: A tile [128 x 32] and B tile [128 x 32] bf16
static __device__ __forceinline__ void copy_tiles(
        const unsigned short* __restrict__ Act, const unsigned short* __restrict__ Wc,
        unsigned short* shA, unsigned short* shB,
        int mBlk, int nBlk, int k0, int tid) {
    const int ln = tid & 3;          // 16-byte chunk within row
    const int row0 = tid >> 2;       // 64 rows per pass
#pragma unroll
    for (int p = 0; p < 2; ++p) {
        const int row = row0 + p * 64;
        const unsigned short* ga = Act + (size_t)(mBlk + row) * KK + k0 + ln * 8;
        const unsigned short* gb = Wc  + (size_t)(nBlk + row) * KK + k0 + ln * 8;
        unsigned short* la = shA + row * PITCH + ln * 8;
        unsigned short* lb = shB + row * PITCH + ln * 8;
#if USE_ASYNC
        __builtin_amdgcn_global_load_async_to_lds_b128(
            (v4i*)(uintptr_t)ga, (as3_v4i)(uintptr_t)la, 0, 0);
        __builtin_amdgcn_global_load_async_to_lds_b128(
            (v4i*)(uintptr_t)gb, (as3_v4i)(uintptr_t)lb, 0, 0);
#else
        *(uint4*)la = *(const uint4*)ga;
        *(uint4*)lb = *(const uint4*)gb;
#endif
    }
}

// ---------------- per-step gate GEMM: gates[1024,2048] = Act @ Wc^T ----------
// grid = (NN/128, BB/128), block = 256 threads = 8 wave32 waves (2 M x 4 N)
__global__ void __launch_bounds__(256)
lstm_gemm(const unsigned short* __restrict__ Act,   // [B, KK] bf16
          const unsigned short* __restrict__ Wc,    // [NN, KK] bf16
          float* __restrict__ gates) {              // [B, NN] f32
    __shared__ unsigned short shA[2][128 * PITCH];
    __shared__ unsigned short shB[2][128 * PITCH];

    const int tid  = threadIdx.x;
    const int lane = tid & 31;
    const int wid  = tid >> 5;
    const int lr   = lane & 15;
    const int hi   = lane >> 4;

    const int mBlk = blockIdx.y * 128;
    const int nBlk = blockIdx.x * 128;
    const int mW   = (wid & 1) * 64;        // wave M offset in tile
    const int nW   = (wid >> 1) * 32;       // wave N offset in tile

    v8f acc[4][2] = {};

    // prologue: fill buffer 0
    copy_tiles(Act, Wc, shA[0], shB[0], mBlk, nBlk, 0, tid);
#if USE_ASYNC
    __builtin_amdgcn_s_wait_asynccnt(0);
#endif
    __syncthreads();

#pragma unroll 1
    for (int kt = 0; kt < KK / 32; ++kt) {
        // kick off next slab into the other buffer
        if (kt + 1 < KK / 32)
            copy_tiles(Act, Wc, shA[(kt + 1) & 1], shB[(kt + 1) & 1],
                       mBlk, nBlk, (kt + 1) * 32, tid);

        const unsigned short* sA = shA[kt & 1];
        const unsigned short* sB = shB[kt & 1];

        // B fragments: 16 contiguous K per lane (lanes 0-15: K0..15 of col N=lr;
        // lanes 16-31: K16..31)
        v16bf bfr[2];
#pragma unroll
        for (int j = 0; j < 2; ++j) {
            const unsigned short* p = sB + (nW + j * 16 + lr) * PITCH + hi * 16;
            V32 u;
            u.q[0] = *(const uint4*)p;
            u.q[1] = *(const uint4*)(p + 8);
            bfr[j] = u.v;
        }

        // A fragments: lanes 0-15 hold K0..15 of row M=lr; lanes 16-31
        // hold K16..31 -> two 16B chunks at hi*16 and hi*16+8
#pragma unroll
        for (int i = 0; i < 4; ++i) {
            const unsigned short* p = sA + (mW + i * 16 + lr) * PITCH + hi * 16;
            V32 u;
            u.q[0] = *(const uint4*)p;
            u.q[1] = *(const uint4*)(p + 8);
            v16bf afr = u.v;
#pragma unroll
            for (int j = 0; j < 2; ++j) {
                acc[i][j] = __builtin_amdgcn_wmma_f32_16x16x32_bf16(
                    false, afr, false, bfr[j], (short)0, acc[i][j], false, false);
            }
        }

        if (kt + 1 < KK / 32) {
#if USE_ASYNC
            __builtin_amdgcn_s_wait_asynccnt(0);
#endif
            __syncthreads();
        }
    }

    // store D fragments: VGPR r -> M = r (+8 for high half-wave), N = lane%16
#pragma unroll
    for (int i = 0; i < 4; ++i) {
#pragma unroll
        for (int j = 0; j < 2; ++j) {
            const int m0 = mBlk + mW + i * 16 + hi * 8;
            const int n0 = nBlk + nW + j * 16 + lr;
            float* gp = gates + (size_t)m0 * NN + n0;
#pragma unroll
            for (int r = 0; r < 8; ++r) gp[(size_t)r * NN] = acc[i][j][r];
        }
    }
}

// ---------------- fused LSTM cell + output projection ----------------
// grid = B blocks, block = 512 threads (one per hidden unit)
__global__ void __launch_bounds__(512)
lstm_cell(const float* __restrict__ gates, const float* __restrict__ bias,
          float* __restrict__ C, unsigned short* __restrict__ Act,
          const float* __restrict__ Wout, const float* __restrict__ bout,
          float* __restrict__ out, int t) {
    const int b = blockIdx.x;
    const int j = threadIdx.x;
    const float* g = gates + (size_t)b * NN;

    float i_ = sigm (g[j]          + bias[j]);
    float f_ = sigm (g[EE + j]     + bias[EE + j]);
    float g_ = tanhf(g[2 * EE + j] + bias[2 * EE + j]);
    float o_ = sigm (g[3 * EE + j] + bias[3 * EE + j]);

    float c = f_ * C[(size_t)b * HH + j] + i_ * g_;
    float h = o_ * tanhf(c);
    C[(size_t)b * HH + j] = c;
    Act[(size_t)b * KK + EE + j] = f2bf(h);   // h-half of next step's A matrix

    __shared__ float red[512];
    red[j] = h * Wout[j];
    __syncthreads();
#pragma unroll
    for (int s = 256; s > 0; s >>= 1) {
        if (j < s) red[j] += red[j + s];
        __syncthreads();
    }
    if (j == 0) out[(size_t)b * WW + t] = sigm(red[0] + bout[0]);
}

// ---------------- host-side orchestration ----------------
extern "C" void kernel_launch(void* const* d_in, const int* in_sizes, int n_in,
                              void* d_out, int out_size, void* d_ws, size_t ws_size,
                              hipStream_t stream) {
    const float* inp  = (const float*)d_in[0];   // [1024, 128*512]
    const float* Wih  = (const float*)d_in[1];   // [2048, 512]
    const float* Whh  = (const float*)d_in[2];   // [2048, 512]
    const float* bih  = (const float*)d_in[3];   // [2048]
    const float* bhh  = (const float*)d_in[4];   // [2048]
    const float* Wout = (const float*)d_in[5];   // [1, 512]
    const float* bout = (const float*)d_in[6];   // [1]
    float* out = (float*)d_out;                  // [1024, 128]

    char* ws = (char*)d_ws;
    unsigned short* Wc  = (unsigned short*)ws;                          // 4 MB
    float* bias         = (float*)(ws + (4u << 20));                    // 8 KB
    unsigned short* Act = (unsigned short*)(ws + (4u << 20) + (16u << 10)); // 2 MB
    float* C            = (float*)(ws + (6u << 20) + (16u << 10));      // 2 MB
    float* gates        = (float*)(ws + (8u << 20) + (16u << 10));      // 8 MB

    lstm_prep<<<(NN * KK + 255) / 256, 256, 0, stream>>>(Wih, Whh, bih, bhh, Wc, bias);
    lstm_init<<<(BB * HH + 255) / 256, 256, 0, stream>>>(C, Act);

    for (int t = 0; t < WW; ++t) {
        lstm_xcast<<<(BB * EE / 8 + 255) / 256, 256, 0, stream>>>(inp, Act, t);
        dim3 g(NN / 128, BB / 128);
        lstm_gemm<<<g, 256, 0, stream>>>(Act, Wc, gates);
        lstm_cell<<<BB, 512, 0, stream>>>(gates, bias, C, Act, Wout, bout, out, t);
    }
}
